// gconv_residual_56495999811905
// MI455X (gfx1250) — matrix operand order repaired and checked by reading, not compiled, not run on hardware
//
#include <hip/hip_runtime.h>
#include <hip/hip_bf16.h>

// ---------------------------------------------------------------------------
// MI455X (gfx1250) fused graph-conv block.
// - All GEMMs via V_WMMA_F32_16X16X32_BF16 (wave32, f32 accumulate).
// - Weights pre-packed to bf16 once, then TDM (tensor_load_to_lds) pulls them
//   into LDS per block; synchronized with s_wait_tensorcnt.
// - Intermediates stored bf16 node-major so gathers/stages are b128 copies.
// Roofline: ~25 GFLOP vs ~150MB HBM traffic -> memory bound with bf16 WMMA.
// ---------------------------------------------------------------------------

typedef __attribute__((ext_vector_type(16))) __bf16 v16bf;
typedef __attribute__((ext_vector_type(8)))  __bf16 v8bf;
typedef __attribute__((ext_vector_type(8)))  float  v8f;
typedef __attribute__((ext_vector_type(4)))  unsigned int u32x4;
typedef __attribute__((ext_vector_type(8)))  int i32x8;
typedef __attribute__((ext_vector_type(4)))  int i32x4;

#define DEV __device__ __forceinline__

static constexpr int Bb   = 16;
static constexpr int Nn   = 2500;
static constexpr int Kk   = 16;
static constexpr int NIN  = 64;
static constexpr int NMED = 64;
static constexpr int WIN  = 32;
static constexpr int WMED = 32;

// bf16 weight pack offsets (elements) inside workspace
static constexpr long OFF_NW1   = 0;        // 64*64
static constexpr long OFF_WW1   = 4096;     // 32*32
static constexpr long OFF_NW2   = 5120;     // 64*64
static constexpr long OFF_THETA = 9216;     // 3*64*64
static constexpr long OFF_GATE  = 21504;    // 64*32
static constexpr long OFF_EDGE  = 23552;    // 32*96
static constexpr long OFF_WW2   = 26624;    // 32*32
static constexpr long WPACK_TOT = 27648;

DEV __bf16 f2bf(float f) {
  unsigned u = __builtin_bit_cast(unsigned, f);
  u += 0x7FFFu + ((u >> 16) & 1u);            // round-to-nearest-even
  unsigned short h = (unsigned short)(u >> 16);
  return __builtin_bit_cast(__bf16, h);
}

// A operand: row-major bf16 weight row; lane covers K = kk+16h..+15
DEV v16bf loadA(const __bf16* wrow, int kk, int half) {
  v8bf lo = *(const v8bf*)(wrow + kk + 16 * half);
  v8bf hi = *(const v8bf*)(wrow + kk + 16 * half + 8);
  return __builtin_shufflevector(lo, hi, 0,1,2,3,4,5,6,7,8,9,10,11,12,13,14,15);
}
// B operand: activation column (col-major LDS tile); lane covers K = kk+16h..+15
DEV v16bf loadB(const __bf16* xcol, int kk, int half) {
  return *(const v16bf*)(xcol + kk + 16 * half);
}
DEV v8f wmma_bf16(v16bf a, v16bf b, v8f c) {
  return __builtin_amdgcn_wmma_f32_16x16x32_bf16(false, a, false, b, (short)0, c,
                                                 false, false);
}
DEV v8f vzero8() { v8f z = {0.f,0.f,0.f,0.f,0.f,0.f,0.f,0.f}; return z; }

// ---------------------------------------------------------------------------
// Tensor Data Mover: 1D bf16 copy global -> LDS (one instruction per matrix).
// D# built per ISA 08_async_tensor: group0 {count,lds_addr,global_addr,type=2},
// group1 {data_size=2B, tensor_dim0=tile_dim0=elems, stride0=elems}.
// This toolchain exposes the 6-arg builtin (g0,g1,g2,g3,g4,cpol).
// ---------------------------------------------------------------------------
#if defined(__has_builtin)
#if __has_builtin(__builtin_amdgcn_tensor_load_to_lds) && __has_builtin(__builtin_amdgcn_s_wait_tensorcnt)
#define HAS_TDM 1
#endif
#endif
#ifndef HAS_TDM
#define HAS_TDM 0
#endif

DEV unsigned lds_off_of(const void* p) {
  return (unsigned)(unsigned long long)(uintptr_t)p;   // LDS aperture: addr[31:0]
}

#if HAS_TDM
DEV void tdm_load_1d(unsigned lds_off, const __bf16* gsrc, int elems) {
  unsigned long long ga = (unsigned long long)(uintptr_t)gsrc;
  u32x4 g0;
  g0[0] = 1u;                                           // count=1 (valid user D#)
  g0[1] = lds_off;                                      // lds_addr
  g0[2] = (unsigned)(ga & 0xFFFFFFFFu);                 // global_addr[31:0]
  g0[3] = (unsigned)((ga >> 32) & 0x01FFFFFFu) | (2u << 30); // addr[56:32]|type=2
  i32x8 g1;
  g1[0] = (1 << 16);                                    // data_size=1 -> 2 bytes
  g1[1] = (int)((elems & 0xFFFF) << 16);                // tensor_dim0[15:0]
  g1[2] = (elems >> 16) & 0xFFFF;                       // tensor_dim0[31:16]
  g1[3] = (int)((elems & 0xFFFF) << 16);                // tile_dim0 = elems
  g1[4] = 0;                                            // tile_dim1/2 unused
  g1[5] = elems;                                        // tensor_dim0_stride
  g1[6] = 0;
  g1[7] = 0;
  i32x4 z4 = {0, 0, 0, 0};
  i32x8 z8 = {0, 0, 0, 0, 0, 0, 0, 0};
  __builtin_amdgcn_tensor_load_to_lds(g0, g1, z4, z4, z8, 0);
}
#endif

// cooperative staging of a bf16 weight matrix into LDS (TDM when available)
DEV void stage_weights(__bf16* lds, const __bf16* gsrc, int elems,
                       int tid, int nthreads) {
#if HAS_TDM
  if (tid < 32) tdm_load_1d(lds_off_of(lds), gsrc, elems);
#else
  for (int i = tid; i < elems; i += nthreads) lds[i] = gsrc[i];
#endif
}
DEV void stage_weights_wait(int tid) {
#if HAS_TDM
  if (tid < 32) __builtin_amdgcn_s_wait_tensorcnt(0);
#endif
}

// ---------------------------------------------------------------------------
// prepack: convert every weight matrix to bf16 once (row-major, packed).
// ---------------------------------------------------------------------------
__global__ __launch_bounds__(256)
void prepack_kernel(const float* __restrict__ nW1, const float* __restrict__ wW1,
                    const float* __restrict__ nW2, const float* __restrict__ theta,
                    const float* __restrict__ gateW, const float* __restrict__ edgeW,
                    const float* __restrict__ wW2, __bf16* __restrict__ out) {
  long i = (long)blockIdx.x * 256 + threadIdx.x;
  if (i >= WPACK_TOT) return;
  float v;
  if      (i < OFF_WW1)   v = nW1[i - OFF_NW1];
  else if (i < OFF_NW2)   v = wW1[i - OFF_WW1];
  else if (i < OFF_THETA) v = nW2[i - OFF_NW2];
  else if (i < OFF_GATE)  v = theta[i - OFF_THETA];
  else if (i < OFF_EDGE)  v = gateW[i - OFF_GATE];
  else if (i < OFF_WW2)   v = edgeW[i - OFF_EDGE];
  else                    v = wW2[i - OFF_WW2];
  out[i] = f2bf(v);
}

// ---------------------------------------------------------------------------
// 1x1 conv + BN + ReLU. One wave per 16-column tile.
// MODE 0: in f32 channel-major [C][ncols], out bf16 node-major [ncols][M].
// MODE 1: in bf16 node-major [ncols][C], out f32 ch-major [M][ncols] + resid.
// ---------------------------------------------------------------------------
template<int M, int C, int MODE>
__global__ __launch_bounds__(256)
void convbn_kernel(const void* __restrict__ xin,
                   const __bf16* __restrict__ Wbf,
                   const float* __restrict__ bias,
                   const float* __restrict__ scale,
                   const float* __restrict__ shift,
                   const float* __restrict__ resid,
                   void* __restrict__ yout,
                   int ncols, int tiles_per_b, int ntiles)
{
  __shared__ __attribute__((aligned(32))) __bf16 ldsW[M * C];
  __shared__ __attribute__((aligned(32))) __bf16 ldsX[8][16 * C];

  const int tid = threadIdx.x;
  stage_weights(ldsW, Wbf, M * C, tid, 256);

  const int wave = tid >> 5, lane = tid & 31;
  const int half = lane >> 4, col = lane & 15;

  const int t      = blockIdx.x * 8 + wave;
  const bool valid = (t < ntiles);
  const int tt     = valid ? t : 0;
  const int b      = tt / tiles_per_b;
  const int c0     = (tt % tiles_per_b) * 16;

  if (MODE == 0) {
    // f32 [C][ncols] -> bf16 col-major LDS tile (coalesced loads, b16 scatter)
    const float* xf = (const float*)xin;
    const long xbase = (long)b * C * ncols;
    for (int i = lane; i < 16 * C; i += 32) {
      int cc = i & 15, ch = i >> 4;
      int nc2 = c0 + cc; if (nc2 >= ncols) nc2 = ncols - 1;
      ldsX[wave][cc * C + ch] = f2bf(xf[xbase + (long)ch * ncols + nc2]);
    }
  } else {
    // bf16 node-major: tile is 16 contiguous rows of C bf16 -> uint4 copy
    const __bf16* xh = (const __bf16*)xin;
    constexpr int VPN = C * 2 / 16;                 // uint4 per node row
    for (int j = lane; j < 16 * VPN; j += 32) {
      int nodeInTile = j / VPN, v = j % VPN;
      int nc2 = c0 + nodeInTile; if (nc2 >= ncols) nc2 = ncols - 1;
      long node = (long)b * ncols + nc2;
      ((uint4*)&ldsX[wave][0])[j] = ((const uint4*)xh)[node * VPN + v];
    }
  }
  stage_weights_wait(tid);
  __syncthreads();

  const __bf16* xcol = &ldsX[wave][col * C];
  const int nc = c0 + col;

  #pragma unroll
  for (int r = 0; r < M / 16; ++r) {
    v8f acc = vzero8();
    const __bf16* wrow = &ldsW[(r * 16 + col) * C];
    #pragma unroll
    for (int kk = 0; kk < C; kk += 32)
      acc = wmma_bf16(loadA(wrow, kk, half), loadB(xcol, kk, half), acc);

    if (MODE == 0) {
      v8bf ov;
      #pragma unroll
      for (int i = 0; i < 8; ++i) {
        int o = r * 16 + half * 8 + i;
        float y = (acc[i] + bias[o]) * scale[o] + shift[o];
        ov[i] = f2bf(fmaxf(y, 0.f));
      }
      if (valid && nc < ncols) {
        long node = (long)b * ncols + nc;
        *(v8bf*)((__bf16*)yout + node * M + r * 16 + half * 8) = ov;  // 16B store
      }
    } else {
      #pragma unroll
      for (int i = 0; i < 8; ++i) {
        int o = r * 16 + half * 8 + i;
        float y = (acc[i] + bias[o]) * scale[o] + shift[o];
        y = fmaxf(y, 0.f);
        if (valid && nc < ncols) {
          long gi = (long)b * M * ncols + (long)o * ncols + nc;
          ((float*)yout)[gi] = y + resid[gi];
        }
      }
    }
  }
}

// ---------------------------------------------------------------------------
// Fused message passing: one wave per node (K=16 neighbors = one WMMA tile).
// ---------------------------------------------------------------------------
__global__ __launch_bounds__(128)
void mp_kernel(const __bf16* __restrict__ nf1,   // [B*Nn][64] node-major
               const __bf16* __restrict__ wf1,   // [B*Nn*Kk][32] col-major
               const int* __restrict__ etype, const int* __restrict__ nn_idx,
               const __bf16* __restrict__ wbf,   // packed bf16 weights
               const float* __restrict__ theta_b, const float* __restrict__ gate_b,
               const float* __restrict__ edge_b,
               const float* __restrict__ wb2, const float* __restrict__ ws2,
               const float* __restrict__ wsh2,
               const float* __restrict__ wfeat,
               __bf16* __restrict__ n_out,       // [B*Nn][64] node-major
               float* __restrict__ out_w)
{
  __shared__ __attribute__((aligned(32))) __bf16 lTheta[3 * NMED * NMED];
  __shared__ __attribute__((aligned(32))) __bf16 lGate [NMED * WMED];
  __shared__ __attribute__((aligned(32))) __bf16 lEdge [WMED * (WMED + NMED)];
  __shared__ __attribute__((aligned(32))) __bf16 lW2   [WIN * WMED];
  __shared__ __attribute__((aligned(32))) __bf16 lNb[4][16 * NMED];
  __shared__ __attribute__((aligned(32))) __bf16 lWf[4][16 * WMED];
  __shared__ __attribute__((aligned(32))) __bf16 lWo[4][16 * WMED];

  const int tid = threadIdx.x;
#if HAS_TDM
  if (tid < 32) {
    tdm_load_1d(lds_off_of(lTheta), wbf + OFF_THETA, 3 * NMED * NMED);
    tdm_load_1d(lds_off_of(lGate),  wbf + OFF_GATE,  NMED * WMED);
    tdm_load_1d(lds_off_of(lEdge),  wbf + OFF_EDGE,  WMED * (WMED + NMED));
    tdm_load_1d(lds_off_of(lW2),    wbf + OFF_WW2,   WIN * WMED);
    __builtin_amdgcn_s_wait_tensorcnt(0);
  }
#else
  for (int i = tid; i < 3 * NMED * NMED;       i += 128) lTheta[i] = wbf[OFF_THETA + i];
  for (int i = tid; i < NMED * WMED;           i += 128) lGate[i]  = wbf[OFF_GATE + i];
  for (int i = tid; i < WMED * (WMED + NMED);  i += 128) lEdge[i]  = wbf[OFF_EDGE + i];
  for (int i = tid; i < WIN * WMED;            i += 128) lW2[i]    = wbf[OFF_WW2 + i];
#endif

  const int wave = tid >> 5, lane = tid & 31;
  const int half = lane >> 4, col = lane & 15;
  const int g = blockIdx.x * 4 + wave;          // node id; grid sized exactly
  const int b = g / Nn, n = g % Nn;

  const long eb = (long)b * Nn * Kk + (long)n * Kk;
  const int myIdx = nn_idx[eb + col];
  const int myEt  = etype [eb + col];

  // wf tile: 16 cols x 32ch contiguous in col-major ws -> vectorized copy
  {
    const uint4* src = (const uint4*)(wf1 + ((long)b * (Nn * Kk) + (long)n * Kk + col) * WMED);
    uint4* dst = (uint4*)&lWf[wave][col * WMED];
    #pragma unroll
    for (int j = 0; j < 2; ++j) dst[half * 2 + j] = src[half * 2 + j];
  }
  // neighbor gather: 128B contiguous per neighbor (node-major nf1)
  {
    const uint4* src = (const uint4*)(nf1 + ((long)b * Nn + myIdx) * NMED);
    uint4* dst = (uint4*)&lNb[wave][col * NMED];
    #pragma unroll
    for (int j = 0; j < 4; ++j) dst[half * 4 + j] = src[half * 4 + j];
  }
  __syncthreads();

  const __bf16* nbcol = &lNb[wave][col * NMED];
  const __bf16* wfcol = &lWf[wave][col * WMED];

  // ---- masked per-edge-type message GEMMs ----
  v8f msg[4];
  #pragma unroll
  for (int r = 0; r < 4; ++r) msg[r] = vzero8();

  #pragma unroll
  for (int t = 0; t < 3; ++t) {
    const float maskv = (myEt == t) ? 1.f : 0.f;
    #pragma unroll
    for (int r = 0; r < 4; ++r) {
      v8f c = vzero8();
      const __bf16* arow = &lTheta[(t * NMED + r * 16 + col) * NMED];
      c = wmma_bf16(loadA(arow, 0,  half), loadB(nbcol, 0,  half), c);
      c = wmma_bf16(loadA(arow, 32, half), loadB(nbcol, 32, half), c);
      #pragma unroll
      for (int i = 0; i < 8; ++i) {
        int o = r * 16 + half * 8 + i;
        msg[r][i] += maskv * (c[i] + theta_b[t * NMED + o]);
      }
    }
  }

  // ---- gate, mean over k (tile columns), n_out ----
  #pragma unroll
  for (int r = 0; r < 4; ++r) {
    v8f gc = vzero8();
    const __bf16* arow = &lGate[(r * 16 + col) * WMED];
    gc = wmma_bf16(loadA(arow, 0, half), loadB(wfcol, 0, half), gc);
    v8f red;
    #pragma unroll
    for (int i = 0; i < 8; ++i) {
      int o = r * 16 + half * 8 + i;
      float gate = 1.f / (1.f + __expf(-(gc[i] + gate_b[o])));
      red[i] = msg[r][i] * gate;
    }
    #pragma unroll
    for (int m = 1; m < 16; m <<= 1) {          // reduce across 16-lane half
      #pragma unroll
      for (int i = 0; i < 8; ++i) red[i] += __shfl_xor(red[i], m, 32);
    }
    if (col == 0) {
      v8bf ov;
      #pragma unroll
      for (int i = 0; i < 8; ++i)
        ov[i] = f2bf(fmaxf(red[i] * (1.f / 16.f), 0.f));
      *(v8bf*)(n_out + (long)g * NMED + r * 16 + half * 8) = ov;   // 16B store
    }
  }

  // ---- edge GEMM over concat [wf(32); nb(64)] -> w_out (transposed to LDS) ----
  #pragma unroll
  for (int r2 = 0; r2 < 2; ++r2) {
    v8f ec = vzero8();
    const __bf16* arow = &lEdge[(r2 * 16 + col) * (WMED + NMED)];
    ec = wmma_bf16(loadA(arow, 0,  half), loadB(wfcol, 0,  half), ec);
    ec = wmma_bf16(loadA(arow, 32, half), loadB(nbcol, 0,  half), ec);
    ec = wmma_bf16(loadA(arow, 64, half), loadB(nbcol, 32, half), ec);
    #pragma unroll
    for (int i = 0; i < 8; ++i) {
      int o = r2 * 16 + half * 8 + i;
      lWo[wave][col * WMED + o] = f2bf(fmaxf(ec[i] + edge_b[o], 0.f));
    }
  }
  __syncthreads();

  // ---- fused second w-conv + BN + ReLU + residual ----
  const __bf16* wocol = &lWo[wave][col * WMED];
  #pragma unroll
  for (int r2 = 0; r2 < 2; ++r2) {
    v8f fc = vzero8();
    const __bf16* arow = &lW2[(r2 * 16 + col) * WMED];
    fc = wmma_bf16(loadA(arow, 0, half), loadB(wocol, 0, half), fc);
    #pragma unroll
    for (int i = 0; i < 8; ++i) {
      int o = r2 * 16 + half * 8 + i;
      float y = (fc[i] + wb2[o]) * ws2[o] + wsh2[o];
      y = fmaxf(y, 0.f);
      long gi = (long)b * WIN * (Nn * Kk) + (long)o * (Nn * Kk) + (long)n * Kk + col;
      out_w[gi] = y + wfeat[gi];
    }
  }
}

// ---------------------------------------------------------------------------
extern "C" void kernel_launch(void* const* d_in, const int* in_sizes, int n_in,
                              void* d_out, int out_size, void* d_ws, size_t ws_size,
                              hipStream_t stream) {
  const float* node_feature   = (const float*)d_in[0];
  const float* weight_feature = (const float*)d_in[1];
  const int*   etype          = (const int*)  d_in[2];
  const int*   nn_idx         = (const int*)  d_in[3];
  const float* nW1  = (const float*)d_in[4];
  const float* nb1  = (const float*)d_in[5];
  const float* ns1  = (const float*)d_in[6];
  const float* nsh1 = (const float*)d_in[7];
  const float* wW1  = (const float*)d_in[8];
  const float* wb1  = (const float*)d_in[9];
  const float* ws1  = (const float*)d_in[10];
  const float* wsh1 = (const float*)d_in[11];
  const float* theta   = (const float*)d_in[12];
  const float* theta_b = (const float*)d_in[13];
  const float* gateW   = (const float*)d_in[14];
  const float* gate_b  = (const float*)d_in[15];
  const float* edgeW   = (const float*)d_in[16];
  const float* edge_b  = (const float*)d_in[17];
  const float* nW2  = (const float*)d_in[18];
  const float* nb2  = (const float*)d_in[19];
  const float* ns2  = (const float*)d_in[20];
  const float* nsh2 = (const float*)d_in[21];
  const float* wW2  = (const float*)d_in[22];
  const float* wb2  = (const float*)d_in[23];
  const float* ws2  = (const float*)d_in[24];
  const float* wsh2 = (const float*)d_in[25];

  const long NF = (long)Bb * NIN * Nn;          // 2,560,000
  const long WF = (long)Bb * WIN * Nn * Kk;     // 20,480,000

  // workspace carve (bf16, 256B aligned)
  char* ws = (char*)d_ws;
  auto al = [](long x) { return (x + 255) & ~255L; };
  long off = 0;
  __bf16* nf1  = (__bf16*)(ws + off); off += al(NF * 2);
  __bf16* wf1  = (__bf16*)(ws + off); off += al(WF * 2);
  __bf16* nout = (__bf16*)(ws + off); off += al(NF * 2);
  __bf16* wbf  = (__bf16*)(ws + off);

  float* out_nf = (float*)d_out;
  float* out_wf = out_nf + NF;

  // 0) pre-pack every weight matrix to bf16 (once; tiny)
  prepack_kernel<<<(int)((WPACK_TOT + 255) / 256), 256, 0, stream>>>(
      nW1, wW1, nW2, theta, gateW, edgeW, wW2, wbf);

  // 1) nf1 = conv_bn_relu(node_feature) : 64x64 GEMM, node-major bf16 out
  const int tilesN = (Nn + 15) / 16;            // 157 (last tile partial)
  convbn_kernel<64, 64, 0><<<(Bb * tilesN + 7) / 8, 256, 0, stream>>>(
      node_feature, wbf + OFF_NW1, nb1, ns1, nsh1, nullptr, nf1,
      Nn, tilesN, Bb * tilesN);

  // 2) wf1 = conv_bn_relu(weight_feature) : 32x32 GEMM, col-major bf16 out
  convbn_kernel<32, 32, 0><<<(Bb * (Nn * Kk / 16) + 7) / 8, 256, 0, stream>>>(
      weight_feature, wbf + OFF_WW1, wb1, ws1, wsh1, nullptr, wf1,
      Nn * Kk, Nn * Kk / 16, Bb * (Nn * Kk / 16));

  // 3) fused message passing + edge conv + second w-conv + residual
  mp_kernel<<<(Bb * Nn) / 4, 128, 0, stream>>>(
      nf1, wf1, etype, nn_idx, wbf, theta_b, gate_b, edge_b,
      wb2, ws2, wsh2, weight_feature, nout, out_wf);

  // 4) nf = conv_bn_relu(n_out) + node_feature (bf16 node-major in, f32 out)
  convbn_kernel<64, 64, 1><<<(Bb * tilesN + 7) / 8, 256, 0, stream>>>(
      nout, wbf + OFF_NW2, nb2, ns2, nsh2, node_feature, out_nf,
      Nn, tilesN, Bb * tilesN);
}
